// Dust3RBlock_4973572129463
// MI455X (gfx1250) — matrix operand order repaired and checked by reading, not compiled, not run on hardware
//
#include <hip/hip_runtime.h>
#include <hip/hip_bf16.h>
#include <math.h>
#include <stdint.h>

// ---------------------------------------------------------------------------
// Dust3R transformer block for MI455X (gfx1250):
//   wave32, WMMA bf16 (v_wmma_f32_16x16x32_bf16), TDM tensor_load_to_lds
//   double-buffered tiles, ds_swizzle softmax reductions.
// ---------------------------------------------------------------------------

typedef __attribute__((ext_vector_type(16))) __bf16 v16bf;
typedef __attribute__((ext_vector_type(8)))  __bf16 v8bf;
typedef __attribute__((ext_vector_type(8)))  float  v8f;
typedef __attribute__((ext_vector_type(4)))  unsigned int v4u;
typedef __attribute__((ext_vector_type(8)))  int v8i;
typedef __attribute__((ext_vector_type(4)))  int v4i;

#define B_   2
#define N_   3072
#define C_   768
#define H_   12
#define HD_  64
#define BN_  (B_ * N_)

#if defined(__HIP_DEVICE_COMPILE__) && __has_builtin(__builtin_amdgcn_tensor_load_to_lds)
#define HAVE_TDM 1
#else
#define HAVE_TDM 0
#endif

union V16U { v16bf v; v8bf h[2]; };

static __device__ __forceinline__ v16bf make16(v8bf lo, v8bf hi) {
  V16U u; u.h[0] = lo; u.h[1] = hi; return u.v;
}

static __device__ __forceinline__ v8f wmma_bf16(v16bf a, v16bf b, v8f c) {
  return __builtin_amdgcn_wmma_f32_16x16x32_bf16(
      false, a, false, b, (short)0, c, false, false);
}

// ds_swizzle xor-lane reduction step (group-of-32 mode, stays inside 16-lane
// halves for masks < 16) -- no LDS memory traffic, no address VGPRs.
template <int MASK>
static __device__ __forceinline__ float swz_xor(float x) {
  return __int_as_float(
      __builtin_amdgcn_ds_swizzle(__float_as_int(x), (MASK << 10) | 0x1f));
}

static __device__ __forceinline__ unsigned lds_off_of(const void* p) {
  // flat LDS address: low 32 bits are the LDS byte address
  return (unsigned)(uintptr_t)p;
}

#if HAVE_TDM
// TDM: DMA a 2D tile (tile_d0 x tile_d1 bf16 elems, global row stride
// stride_elems) into LDS, inserting pad_amount DWORDs of padding every
// (1<<pad_iv)*2 DWORDs (codes per D# spec) to create a padded LDS stride.
static __device__ __forceinline__ void tdm_load_2d(
    const void* gptr, unsigned lds_byte, unsigned tile_d0, unsigned tile_d1,
    unsigned long long stride_elems, unsigned pad_iv, unsigned pad_amt) {
  const unsigned long long ga = (unsigned long long)(uintptr_t)gptr;
  v4u g0;
  g0.x = 1u;                                       // count=1, user descriptor
  g0.y = lds_byte;                                 // lds_addr
  g0.z = (unsigned)(ga & 0xffffffffull);           // global_addr[31:0]
  g0.w = (unsigned)((ga >> 32) & 0x01ffffffull)    // global_addr[56:32]
         | (2u << 30);                             // type=2 (image)
  v8i g1;
  g1[0] = (int)((1u << 16)                         // data_size = 2 bytes
              | (1u << 20)                         // pad_enable
              | (pad_iv << 22) | (pad_amt << 25)); // pad codes
  g1[1] = (int)((tile_d0 & 0xffffu) << 16);        // tensor_dim0[15:0]
  g1[2] = (int)(((tile_d0 >> 16) & 0xffffu) | ((tile_d1 & 0xffffu) << 16));
  g1[3] = (int)(((tile_d1 >> 16) & 0xffffu) | ((tile_d0 & 0xffffu) << 16));
  g1[4] = (int)(tile_d1 & 0xffffu);                // tile_dim1 (tile_dim2=0)
  g1[5] = (int)(stride_elems & 0xffffffffull);     // tensor_dim0_stride[31:0]
  g1[6] = (int)((stride_elems >> 32) & 0xffffull); // stride[47:32]
  g1[7] = 0;
  v4i g2 = {1, 1, 0, 0};                           // dim2=1, dim3=1
  v4i g3 = {0, (int)(1u << 16), 0, 0};             // dim4=1
#if defined(__clang_major__) && __clang_major__ >= 23
  v8i gz = {0, 0, 0, 0, 0, 0, 0, 0};
  __builtin_amdgcn_tensor_load_to_lds(g0, g1, g2, g3, gz, 0);
#else
  __builtin_amdgcn_tensor_load_to_lds(g0, g1, g2, g3, 0);
#endif
}
#endif

// ---------------------------------------------------------------------------
// f32 -> bf16 conversion (weights)
// ---------------------------------------------------------------------------
__global__ __launch_bounds__(256)
void cvt_bf16_kernel(const float* __restrict__ in, __bf16* __restrict__ out, int n) {
  int i = blockIdx.x * 256 + threadIdx.x;
  if (i < n) out[i] = (__bf16)in[i];
}

// ---------------------------------------------------------------------------
// LayerNorm over C=768, one block per row; writes bf16 normalized output.
// ---------------------------------------------------------------------------
__global__ __launch_bounds__(256)
void layernorm_kernel(const float* __restrict__ x, const float* __restrict__ g,
                      const float* __restrict__ be, __bf16* __restrict__ out) {
  __shared__ float red[256];
  const int row = blockIdx.x;
  const int t = threadIdx.x;
  const float* xr = x + (size_t)row * C_;
  float v[3];
  float s = 0.0f;
#pragma unroll
  for (int i = 0; i < 3; ++i) { v[i] = xr[t + i * 256]; s += v[i]; }
  red[t] = s;
  __syncthreads();
  for (int off = 128; off > 0; off >>= 1) {
    if (t < off) red[t] += red[t + off];
    __syncthreads();
  }
  const float mean = red[0] * (1.0f / (float)C_);
  __syncthreads();
  float sq = 0.0f;
#pragma unroll
  for (int i = 0; i < 3; ++i) { float d = v[i] - mean; sq += d * d; }
  red[t] = sq;
  __syncthreads();
  for (int off = 128; off > 0; off >>= 1) {
    if (t < off) red[t] += red[t + off];
    __syncthreads();
  }
  const float rinv = rsqrtf(red[0] * (1.0f / (float)C_) + 1e-5f);
#pragma unroll
  for (int i = 0; i < 3; ++i) {
    const int c = t + i * 256;
    out[(size_t)row * C_ + c] = (__bf16)((v[i] - mean) * rinv * g[c] + be[c]);
  }
}

// ---------------------------------------------------------------------------
// WMMA GEMM: D[M,N] = A_bf16[M,K] * W_bf16[N,K]^T (+ epilogue)
//   EPI 0: store bf16 | EPI 1: +bias+resid -> f32 | EPI 2: gelu(+bias) -> bf16
// Block tile 128x128, 8 waves (2x4), wave tile 64x32, K-step 32.
// TDM double-buffered LDS staging (padded stride 40 elems via D# pad fields).
// ---------------------------------------------------------------------------
template <int EPI>
__global__ __launch_bounds__(256)
void gemm_wmma_kernel(const __bf16* __restrict__ A, const __bf16* __restrict__ W,
                      const float* __restrict__ bias, const float* __restrict__ resid,
                      float* __restrict__ outF, __bf16* __restrict__ outB,
                      int M, int Nout, int K) {
  __shared__ __bf16 lA[2][128][40];
  __shared__ __bf16 lB[2][128][40];
  const int bm = blockIdx.y * 128;
  const int bn = blockIdx.x * 128;
  const int t = threadIdx.x;
  const int lane = t & 31;
  const int wave = t >> 5;
  const int wm = (wave >> 2) * 64;
  const int wn = (wave & 3) * 32;
  const int lr = lane & 15;
  const int hi = (lane >> 4) & 1;
  const int nkt = K >> 5;

  const v8f vzero = {0.f, 0.f, 0.f, 0.f, 0.f, 0.f, 0.f, 0.f};
  v8f acc[4][2];
#pragma unroll
  for (int mt = 0; mt < 4; ++mt)
#pragma unroll
    for (int nt = 0; nt < 2; ++nt) acc[mt][nt] = vzero;

#if HAVE_TDM
  if (__builtin_amdgcn_readfirstlane(t) < 32) {   // wave 0 (scalar branch)
    tdm_load_2d(A + (size_t)bm * K, lds_off_of(&lA[0][0][0]), 32, 128,
                (unsigned long long)K, 3, 3);     // 16dw rows, pad 4dw -> 40el
    tdm_load_2d(W + (size_t)bn * K, lds_off_of(&lB[0][0][0]), 32, 128,
                (unsigned long long)K, 3, 3);
  }
#else
  const int ldrow = t >> 1;
  const int ldcol = (t & 1) << 4;
#endif

  for (int it = 0; it < nkt; ++it) {
    const int buf = it & 1;
#if HAVE_TDM
    if (__builtin_amdgcn_readfirstlane(t) < 32) {
      if (it + 1 < nkt) {                          // DMA next tile, wait current
        tdm_load_2d(A + (size_t)bm * K + (size_t)(it + 1) * 32,
                    lds_off_of(&lA[buf ^ 1][0][0]), 32, 128,
                    (unsigned long long)K, 3, 3);
        tdm_load_2d(W + (size_t)bn * K + (size_t)(it + 1) * 32,
                    lds_off_of(&lB[buf ^ 1][0][0]), 32, 128,
                    (unsigned long long)K, 3, 3);
        __builtin_amdgcn_s_wait_tensorcnt(2);
      } else {
        __builtin_amdgcn_s_wait_tensorcnt(0);
      }
    }
    __syncthreads();                               // current buf visible to all
#else
    {
      const __bf16* ga = A + (size_t)(bm + ldrow) * K + it * 32 + ldcol;
      const __bf16* gw = W + (size_t)(bn + ldrow) * K + it * 32 + ldcol;
      v8bf a0 = *(const v8bf*)(ga);
      v8bf a1 = *(const v8bf*)(ga + 8);
      v8bf w0 = *(const v8bf*)(gw);
      v8bf w1 = *(const v8bf*)(gw + 8);
      __syncthreads();
      *(v8bf*)&lA[buf][ldrow][ldcol]     = a0;
      *(v8bf*)&lA[buf][ldrow][ldcol + 8] = a1;
      *(v8bf*)&lB[buf][ldrow][ldcol]     = w0;
      *(v8bf*)&lB[buf][ldrow][ldcol + 8] = w1;
      __syncthreads();
    }
#endif

    v16bf af[4], bfrag[2];
#pragma unroll
    for (int mt = 0; mt < 4; ++mt) {
      const __bf16* ar = &lA[buf][wm + mt * 16 + lr][0];
      af[mt] = make16(*(const v8bf*)&ar[hi ? 8 : 0],
                      *(const v8bf*)&ar[hi ? 24 : 16]);
    }
#pragma unroll
    for (int nt = 0; nt < 2; ++nt) {
      const __bf16* br = &lB[buf][wn + nt * 16 + lr][0];
      bfrag[nt] = make16(*(const v8bf*)&br[hi ? 8 : 0], *(const v8bf*)&br[hi ? 24 : 16]);
    }
#pragma unroll
    for (int mt = 0; mt < 4; ++mt)
#pragma unroll
      for (int nt = 0; nt < 2; ++nt)
        acc[mt][nt] = wmma_bf16(af[mt], bfrag[nt], acc[mt][nt]);

#if HAVE_TDM
    __syncthreads();   // release buf^1 before next iteration's TDM write
#endif
  }

  // Epilogue (C/D layout: VGPR r -> rows r / r+8)
#pragma unroll
  for (int mt = 0; mt < 4; ++mt) {
#pragma unroll
    for (int nt = 0; nt < 2; ++nt) {
#pragma unroll
      for (int r = 0; r < 8; ++r) {
        const int gm = bm + wm + mt * 16 + r + (hi ? 8 : 0);
        const int gn = bn + wn + nt * 16 + lr;
        float v = acc[mt][nt][r];
        if (EPI == 0) {
          outB[(size_t)gm * Nout + gn] = (__bf16)v;
        } else if (EPI == 1) {
          v += bias[gn] + resid[(size_t)gm * Nout + gn];
          outF[(size_t)gm * Nout + gn] = v;
        } else {
          v += bias[gn];
          const float gl = 0.5f * v * (1.0f + erff(v * 0.70710678118654752f));
          outB[(size_t)gm * Nout + gn] = (__bf16)gl;
        }
      }
    }
  }
}

// ---------------------------------------------------------------------------
// 2D RoPE on q,k (+ v passthrough), qkv[BN,3C] bf16 -> q/k/v [B,H,N,HD] bf16
// ---------------------------------------------------------------------------
__global__ __launch_bounds__(64)
void rope_kernel(const __bf16* __restrict__ qkv,
                 const float* __restrict__ cosx, const float* __restrict__ sinx,
                 const float* __restrict__ cosy, const float* __restrict__ siny,
                 __bf16* __restrict__ Q, __bf16* __restrict__ Ko,
                 __bf16* __restrict__ V) {
  const int idx = blockIdx.x;            // b*H*N + h*N + n
  const int n = idx % N_;
  const int h = (idx / N_) % H_;
  const int b = idx / (N_ * H_);
  const int e = threadIdx.x;             // 0..63
  const size_t row = ((size_t)b * N_ + n) * (3 * C_);
  const int coff = h * HD_ + e;
  const float q = (float)qkv[row + 0 * C_ + coff];
  const float k = (float)qkv[row + 1 * C_ + coff];
  const float v = (float)qkv[row + 2 * C_ + coff];
  const int half = e >> 5;               // 0: y-rope, 1: x-rope
  const int w = e & 31;
  const float* ct = half ? cosx : cosy;
  const float* st = half ? sinx : siny;
  const float c = ct[n * 32 + w];
  const float s = st[n * 32 + w];
  const int pe = (w < 16) ? (e + 16) : (e - 16);   // rotate-half partner
  const float qp = (float)qkv[row + 0 * C_ + h * HD_ + pe];
  const float kp = (float)qkv[row + 1 * C_ + h * HD_ + pe];
  const float sgn = (w < 16) ? -1.0f : 1.0f;
  const size_t orow = (((size_t)b * H_ + h) * N_ + n) * HD_ + e;
  Q[orow]  = (__bf16)(q * c + sgn * qp * s);
  Ko[orow] = (__bf16)(k * c + sgn * kp * s);
  V[orow]  = (__bf16)v;
}

// ---------------------------------------------------------------------------
// Flash attention: grid (N/64 q-tiles, B*H). 4 waves, 16 q-rows per wave.
// K tile: TDM double-buffered. V tile: register-pipelined + LDS transpose.
// S = Q K^T (WMMA), online softmax (ds_swizzle reductions), O += P V (WMMA).
// ---------------------------------------------------------------------------
__global__ __launch_bounds__(128)
void flash_attn_kernel(const __bf16* __restrict__ Q, const __bf16* __restrict__ Kin,
                       const __bf16* __restrict__ Vin, __bf16* __restrict__ O) {
  __shared__ __bf16 kt_s[2][64][72];  // K tile   [key][hd], double buffered
  __shared__ __bf16 vt_s[64][72];     // V^T tile [hd][key]
  __shared__ __bf16 p_s[4][16][72];   // per-wave P staging (C->A layout)
  const int qt = blockIdx.x;
  const int bh = blockIdx.y;
  const int b = bh / H_;
  const int h = bh % H_;
  const int t = threadIdx.x;
  const int lane = t & 31;
  const int wave = t >> 5;
  const int lr = lane & 15;
  const int hi = (lane >> 4) & 1;
  const int NT = N_ / 64;

  const size_t head_off = (size_t)bh * N_ * HD_;

  // Q A-fragments for this wave's 16 rows (K = HD = 64 -> two k-steps)
  const __bf16* qbase = Q + head_off + (size_t)(qt * 64 + wave * 16 + lr) * HD_;
  v16bf qa[2];
#pragma unroll
  for (int ks = 0; ks < 2; ++ks)
    qa[ks] = make16(*(const v8bf*)&qbase[ks * 32 + (hi ? 8 : 0)],
                    *(const v8bf*)&qbase[ks * 32 + (hi ? 24 : 16)]);

  const v8f vzero = {0.f, 0.f, 0.f, 0.f, 0.f, 0.f, 0.f, 0.f};
  float m_run[8], l_run[8];
  v8f oacc[4];
#pragma unroll
  for (int r = 0; r < 8; ++r) { m_run[r] = -1e30f; l_run[r] = 0.0f; }
#pragma unroll
  for (int nt = 0; nt < 4; ++nt) oacc[nt] = vzero;

  const int ldrow = t >> 1;         // 0..63
  const int ldch  = (t & 1) * 32;   // 0 or 32

  // prologue: V tile 0 into registers, K tile 0 via TDM
  v8bf vreg[4];
  {
    const __bf16* vg = Vin + head_off + (size_t)ldrow * HD_ + ldch;
#pragma unroll
    for (int i = 0; i < 4; ++i) vreg[i] = *(const v8bf*)&vg[i * 8];
  }
#if HAVE_TDM
  if (__builtin_amdgcn_readfirstlane(t) < 32) {
    tdm_load_2d(Kin + head_off, lds_off_of(&kt_s[0][0][0]), 64, 64,
                (unsigned long long)HD_, 4, 3);   // 32dw rows, pad 4dw -> 72el
  }
#endif

  for (int kt = 0; kt < NT; ++kt) {
    const int buf = kt & 1;
    __syncthreads();                 // previous compute done reading LDS
    // V transpose from registers into LDS
#pragma unroll
    for (int i = 0; i < 4; ++i)
#pragma unroll
      for (int j = 0; j < 8; ++j)
        vt_s[ldch + i * 8 + j][ldrow] = vreg[i][j];
#if HAVE_TDM
    if (__builtin_amdgcn_readfirstlane(t) < 32) {
      if (kt + 1 < NT) {
        tdm_load_2d(Kin + head_off + (size_t)(kt + 1) * 64 * HD_,
                    lds_off_of(&kt_s[buf ^ 1][0][0]), 64, 64,
                    (unsigned long long)HD_, 4, 3);
        __builtin_amdgcn_s_wait_tensorcnt(1);
      } else {
        __builtin_amdgcn_s_wait_tensorcnt(0);
      }
    }
#else
    {
      const __bf16* kg = Kin + head_off + (size_t)(kt * 64 + ldrow) * HD_ + ldch;
#pragma unroll
      for (int i = 0; i < 4; ++i)
        *(v8bf*)&kt_s[buf][ldrow][ldch + i * 8] = *(const v8bf*)&kg[i * 8];
    }
#endif
    __syncthreads();                 // K tile + V^T visible

    // prefetch next V tile into registers (overlaps WMMA compute)
    if (kt + 1 < NT) {
      const __bf16* vg =
          Vin + head_off + (size_t)((kt + 1) * 64 + ldrow) * HD_ + ldch;
#pragma unroll
      for (int i = 0; i < 4; ++i) vreg[i] = *(const v8bf*)&vg[i * 8];
    }

    // ---- S = (Q K^T) * HD^-0.5 -------------------------------------------
    v8f sc[4];
#pragma unroll
    for (int nt = 0; nt < 4; ++nt) sc[nt] = vzero;
#pragma unroll
    for (int ks = 0; ks < 2; ++ks) {
#pragma unroll
      for (int nt = 0; nt < 4; ++nt) {
        const __bf16* br = &kt_s[buf][nt * 16 + lr][ks * 32];
        v16bf bfr = make16(*(const v8bf*)&br[hi ? 8 : 0], *(const v8bf*)&br[hi ? 24 : 16]);
        sc[nt] = wmma_bf16(qa[ks], bfr, sc[nt]);
      }
    }
#pragma unroll
    for (int nt = 0; nt < 4; ++nt)
#pragma unroll
      for (int r = 0; r < 8; ++r) sc[nt][r] *= 0.125f;

    // ---- online softmax (ds_swizzle xor reductions in 16-lane halves) ----
    float mx[8];
#pragma unroll
    for (int r = 0; r < 8; ++r) {
      float m0 = fmaxf(fmaxf(sc[0][r], sc[1][r]), fmaxf(sc[2][r], sc[3][r]));
      m0 = fmaxf(m0, swz_xor<1>(m0));
      m0 = fmaxf(m0, swz_xor<2>(m0));
      m0 = fmaxf(m0, swz_xor<4>(m0));
      m0 = fmaxf(m0, swz_xor<8>(m0));
      mx[r] = m0;
    }
    float alpha[8], rs[8];
#pragma unroll
    for (int r = 0; r < 8; ++r) {
      const float mn = fmaxf(m_run[r], mx[r]);
      alpha[r] = __expf(m_run[r] - mn);
      m_run[r] = mn;
      rs[r] = 0.0f;
    }
#pragma unroll
    for (int nt = 0; nt < 4; ++nt)
#pragma unroll
      for (int r = 0; r < 8; ++r) {
        const float p = __expf(sc[nt][r] - m_run[r]);
        sc[nt][r] = p;
        rs[r] += p;
      }
#pragma unroll
    for (int r = 0; r < 8; ++r) {
      float s0 = rs[r];
      s0 += swz_xor<1>(s0);
      s0 += swz_xor<2>(s0);
      s0 += swz_xor<4>(s0);
      s0 += swz_xor<8>(s0);
      l_run[r] = l_run[r] * alpha[r] + s0;
    }
#pragma unroll
    for (int nt = 0; nt < 4; ++nt)
#pragma unroll
      for (int r = 0; r < 8; ++r) oacc[nt][r] *= alpha[r];

    // ---- P: C-layout regs -> wave-private LDS -> A-layout fragments ------
#pragma unroll
    for (int nt = 0; nt < 4; ++nt)
#pragma unroll
      for (int r = 0; r < 8; ++r)
        p_s[wave][r + (hi ? 8 : 0)][nt * 16 + lr] = (__bf16)sc[nt][r];
    // same-wave LDS ops are in-order; no block barrier needed for p_s[wave]

    // ---- O += P V --------------------------------------------------------
#pragma unroll
    for (int ks = 0; ks < 2; ++ks) {
      const __bf16* pr = &p_s[wave][lr][ks * 32];
      v16bf pa = make16(*(const v8bf*)&pr[hi ? 8 : 0],
                        *(const v8bf*)&pr[hi ? 24 : 16]);
#pragma unroll
      for (int nt = 0; nt < 4; ++nt) {
        const __bf16* br = &vt_s[nt * 16 + lr][ks * 32];
        v16bf bfr = make16(*(const v8bf*)&br[hi ? 8 : 0], *(const v8bf*)&br[hi ? 24 : 16]);
        oacc[nt] = wmma_bf16(pa, bfr, oacc[nt]);
      }
    }
  }

  // ---- normalize and write O as [B,N,C] bf16 ------------------------------
#pragma unroll
  for (int nt = 0; nt < 4; ++nt)
#pragma unroll
    for (int r = 0; r < 8; ++r) {
      const float v = oacc[nt][r] / l_run[r];
      const int row = qt * 64 + wave * 16 + r + (hi ? 8 : 0);
      const int col = h * HD_ + nt * 16 + lr;
      O[((size_t)b * N_ + row) * C_ + col] = (__bf16)v;
    }
}

// ---------------------------------------------------------------------------
// Launcher
// ---------------------------------------------------------------------------
extern "C" void kernel_launch(void* const* d_in, const int* in_sizes, int n_in,
                              void* d_out, int out_size, void* d_ws, size_t ws_size,
                              hipStream_t stream) {
  const float* x      = (const float*)d_in[0];
  const float* w_qkv  = (const float*)d_in[1];
  const float* w_proj = (const float*)d_in[2];
  const float* b_proj = (const float*)d_in[3];
  const float* g1     = (const float*)d_in[4];
  const float* beta1  = (const float*)d_in[5];
  const float* g2     = (const float*)d_in[6];
  const float* beta2  = (const float*)d_in[7];
  const float* w_fc1  = (const float*)d_in[8];
  const float* b_fc1  = (const float*)d_in[9];
  const float* w_fc2  = (const float*)d_in[10];
  const float* b_fc2  = (const float*)d_in[11];
  const float* cosx   = (const float*)d_in[12];
  const float* sinx   = (const float*)d_in[13];
  const float* cosy   = (const float*)d_in[14];
  const float* siny   = (const float*)d_in[15];
  float* out = (float*)d_out;

  char* p = (char*)d_ws;
  auto carve = [&](size_t bytes) -> char* {
    char* r = p;
    p += (bytes + 255) & ~((size_t)255);
    return r;
  };
  __bf16* xn      = (__bf16*)carve((size_t)BN_ * C_ * 2);
  __bf16* wqkv_b  = (__bf16*)carve((size_t)3 * C_ * C_ * 2);
  __bf16* wproj_b = (__bf16*)carve((size_t)C_ * C_ * 2);
  __bf16* wfc1_b  = (__bf16*)carve((size_t)4 * C_ * C_ * 2);
  __bf16* wfc2_b  = (__bf16*)carve((size_t)C_ * 4 * C_ * 2);
  __bf16* qkv_b   = (__bf16*)carve((size_t)BN_ * 3 * C_ * 2);
  __bf16* qr      = (__bf16*)carve((size_t)B_ * H_ * N_ * HD_ * 2);
  __bf16* kr      = (__bf16*)carve((size_t)B_ * H_ * N_ * HD_ * 2);
  __bf16* vr      = (__bf16*)carve((size_t)B_ * H_ * N_ * HD_ * 2);
  __bf16* o_b     = (__bf16*)carve((size_t)BN_ * C_ * 2);
  float*  x1      = (float*) carve((size_t)BN_ * C_ * 4);
  __bf16* h_b     = (__bf16*)carve((size_t)BN_ * 4 * C_ * 2);

  {
    const int nq = 3 * C_ * C_, np = C_ * C_, nf = 4 * C_ * C_;
    cvt_bf16_kernel<<<(nq + 255) / 256, 256, 0, stream>>>(w_qkv, wqkv_b, nq);
    cvt_bf16_kernel<<<(np + 255) / 256, 256, 0, stream>>>(w_proj, wproj_b, np);
    cvt_bf16_kernel<<<(nf + 255) / 256, 256, 0, stream>>>(w_fc1, wfc1_b, nf);
    cvt_bf16_kernel<<<(nf + 255) / 256, 256, 0, stream>>>(w_fc2, wfc2_b, nf);
  }

  layernorm_kernel<<<BN_, 256, 0, stream>>>(x, g1, beta1, xn);

  gemm_wmma_kernel<0><<<dim3(3 * C_ / 128, BN_ / 128), 256, 0, stream>>>(
      xn, wqkv_b, nullptr, nullptr, nullptr, qkv_b, BN_, 3 * C_, C_);

  rope_kernel<<<B_ * H_ * N_, 64, 0, stream>>>(qkv_b, cosx, sinx, cosy, siny,
                                               qr, kr, vr);

  flash_attn_kernel<<<dim3(N_ / 64, B_ * H_), 128, 0, stream>>>(qr, kr, vr, o_b);

  gemm_wmma_kernel<1><<<dim3(C_ / 128, BN_ / 128), 256, 0, stream>>>(
      o_b, wproj_b, b_proj, x, x1, nullptr, BN_, C_, C_);

  layernorm_kernel<<<BN_, 256, 0, stream>>>(x1, g2, beta2, xn);

  gemm_wmma_kernel<2><<<dim3(4 * C_ / 128, BN_ / 128), 256, 0, stream>>>(
      xn, wfc1_b, b_fc1, nullptr, nullptr, h_b, BN_, 4 * C_, C_);

  gemm_wmma_kernel<1><<<dim3(C_ / 128, BN_ / 128), 256, 0, stream>>>(
      h_b, wfc2_b, b_fc2, x1, out, nullptr, BN_, C_, 4 * C_);
}
